// CrossAttention_35270271435567
// MI455X (gfx1250) — matrix operand order repaired and checked by reading, not compiled, not run on hardware
//
#include <hip/hip_runtime.h>
#include <hip/hip_bf16.h>
#include <stdint.h>

// ---------------------------------------------------------------------------
// CDNA5 (gfx1250) cross-attention: bf16 WMMA GEMMs with f32 accumulate,
// double-buffered LDS fed by GLOBAL_LOAD_ASYNC_TO_LDS (ASYNCcnt-tracked).
// ---------------------------------------------------------------------------

typedef __attribute__((ext_vector_type(16))) __bf16       v16bf;
typedef __attribute__((ext_vector_type(8)))  float        v8f;
typedef __attribute__((ext_vector_type(4)))  unsigned int u32x4;

union FragU { v16bf v; u32x4 q[2]; };

#define BM   256
#define BN   128
#define BK   32
#define LDSS 40   // LDS row stride in bf16 elements (32 + 8 pad, keeps 16B align)

// Async copy of 16 bytes global -> LDS, per lane. The LDS byte address is the
// low 32 bits of the generic pointer (ISA: LDS aperture uses addr[31:0]).
__device__ __forceinline__ void async_b128(const __bf16* g, const __bf16* l)
{
    asm volatile("global_load_async_to_lds_b128 %0, %1, off"
                 :: "v"((unsigned)(uintptr_t)l), "v"(g)
                 : "memory");
}

// C[M,N] = scale * (A[M,K] @ B[N,K]^T) + bias[col]
// A,B bf16 row-major; optional f32 and/or bf16 output; optional transposed
// store (C[col*M + row]) used for V^T and the final (b,c,h,w) output.
template<bool TRANS>
__global__ __launch_bounds__(256)
void wmma_gemm_bf16(const __bf16* __restrict__ A, const __bf16* __restrict__ B,
                    const float* __restrict__ bias,
                    float* __restrict__ Cf, __bf16* __restrict__ Cb,
                    int M, int N, int K,
                    long sA, long sB, long sC, float scale)
{
    __shared__ __align__(16) __bf16 As[2][BM][LDSS];
    __shared__ __align__(16) __bf16 Bs[2][BN][LDSS];

    const int bz = blockIdx.z;
    A += (long)bz * sA;
    B += (long)bz * sB;
    const long cbase = (long)bz * sC;

    const int rowBase = blockIdx.y * BM;
    const int colBase = blockIdx.x * BN;

    const int tid  = threadIdx.x;
    const int lane = tid & 31;
    const int wid  = tid >> 5;      // 8 wave32 waves: 4 (M) x 2 (N)
    const int wm   = wid >> 1;      // 0..3 -> 64 rows each
    const int wn   = wid & 1;       // 0..1 -> 64 cols each
    const int m0   = wm * 64;
    const int n0   = wn * 64;
    const int lr   = lane & 15;
    const int lh   = lane >> 4;

    // Per-thread staging chunks (8 bf16 = 16B each): A tile 256x32 -> 4/thread,
    // B tile 128x32 -> 2/thread.
    int ar[4], akc[4], br[2], bkc[2];
#pragma unroll
    for (int t = 0; t < 4; t++) { int c = tid + t * 256; ar[t] = c >> 2; akc[t] = (c & 3) * 8; }
#pragma unroll
    for (int t = 0; t < 2; t++) { int c = tid + t * 256; br[t] = c >> 2; bkc[t] = (c & 3) * 8; }

    auto issue = [&](int buf, int kt) {
#pragma unroll
        for (int t = 0; t < 4; t++)
            async_b128(&A[(long)(rowBase + ar[t]) * K + kt + akc[t]], &As[buf][ar[t]][akc[t]]);
#pragma unroll
        for (int t = 0; t < 2; t++)
            async_b128(&B[(long)(colBase + br[t]) * K + kt + bkc[t]], &Bs[buf][br[t]][bkc[t]]);
    };

    v8f zero = {};
    v8f acc[4][4];
#pragma unroll
    for (int mi = 0; mi < 4; mi++)
#pragma unroll
        for (int ni = 0; ni < 4; ni++) acc[mi][ni] = zero;

    const int nsteps = K / BK;
    issue(0, 0);

    for (int s = 0; s < nsteps; ++s) {
        const int buf = s & 1;
        if (s + 1 < nsteps) {
            issue(buf ^ 1, (s + 1) * BK);
            // 12 outstanding; wait until only the 6 just-issued remain
            // (async loads complete in order), i.e. current buffer is ready.
            asm volatile("s_wait_asynccnt 0x6" ::: "memory");
        } else {
            asm volatile("s_wait_asynccnt 0x0" ::: "memory");
        }
        __syncthreads();   // current buffer visible to all waves

        // A fragment (ISA 7.12.2, 16-bit A 16x32): lane row M=L%16,
        // K = {(L>>4)*8 .. +7} and {16+(L>>4)*8 .. +7} -> two b128 LDS loads.
        FragU af[4], bfr[4];
#pragma unroll
        for (int mi = 0; mi < 4; mi++) {
            int row = m0 + mi * 16 + lr;
            int kb  = lh * 8;
            af[mi].q[0] = *(const u32x4*)&As[buf][row][kb];
            af[mi].q[1] = *(const u32x4*)&As[buf][row][16 + kb];
        }
        // B fragment: lane column N=L%16, K = {(L>>4)*8 .. +7} and {16+(L>>4)*8 .. +7}.
#pragma unroll
        for (int ni = 0; ni < 4; ni++) {
            int row = n0 + ni * 16 + lr;
            int ks  = lh * 8;
            bfr[ni].q[0] = *(const u32x4*)&Bs[buf][row][ks];
            bfr[ni].q[1] = *(const u32x4*)&Bs[buf][row][16 + ks];
        }

#pragma unroll
        for (int mi = 0; mi < 4; mi++)
#pragma unroll
            for (int ni = 0; ni < 4; ni++)
                acc[mi][ni] = __builtin_amdgcn_wmma_f32_16x16x32_bf16(
                    false, af[mi].v, false, bfr[ni].v,
                    (short)0, acc[mi][ni], false, false);

        __syncthreads();   // everyone done reading buf before s+1 refills it
    }

    // C/D layout: VGPR g <-> row (L>>4)*8 + g, col L%16.
#pragma unroll
    for (int mi = 0; mi < 4; mi++) {
#pragma unroll
        for (int ni = 0; ni < 4; ni++) {
            int colb = colBase + n0 + ni * 16 + lr;
            float bv = bias ? bias[colb] : 0.0f;
#pragma unroll
            for (int g = 0; g < 8; g++) {
                int row = rowBase + m0 + mi * 16 + lh * 8 + g;
                float v = acc[mi][ni][g] * scale + bv;
                long off = TRANS ? (cbase + (long)colb * M + row)
                                 : (cbase + (long)row * N + colb);
                if (Cf) Cf[off] = v;
                if (Cb) Cb[off] = (__bf16)v;
            }
        }
    }
}

// ---------------------------------------------------------------------------

__global__ void k_f32_to_bf16(const float* __restrict__ in, __bf16* __restrict__ out, long n)
{
    long i = (long)blockIdx.x * blockDim.x + threadIdx.x;
    long stride = (long)gridDim.x * blockDim.x;
    for (; i < n; i += stride) out[i] = (__bf16)in[i];
}

// x: (8, 512, 1024 spatial) f32 -> tokens (8, 1024, 512) bf16
__global__ void k_x_to_tokens(const float* __restrict__ x, __bf16* __restrict__ out)
{
    long i = (long)blockIdx.x * blockDim.x + threadIdx.x;
    const long total = 8L * 1024 * 512;
    long stride = (long)gridDim.x * blockDim.x;
    for (; i < total; i += stride) {
        long b   = i >> 19;               // / (1024*512)
        long rem = i & ((1L << 19) - 1);
        long n   = rem >> 9;              // token
        long c   = rem & 511;             // channel
        out[i] = (__bf16)x[((b << 9) + c) * 1024 + n];
    }
}

// Row softmax, in place on f32 scores (S == 1024, block == 256 threads),
// plus bf16 copy of att for the att @ V GEMM.
__global__ __launch_bounds__(256)
void k_softmax(float* __restrict__ att, const uint8_t* __restrict__ mask,
               __bf16* __restrict__ attb, int S)
{
    const int row = blockIdx.x;               // b*1024 + token
    float*         p  = att  + (long)row * S;
    const uint8_t* m  = mask + (long)row * S;
    __bf16*        pb = attb + (long)row * S;
    const int tid = threadIdx.x;

    float v[4];
    float mx = -1e30f;
#pragma unroll
    for (int t = 0; t < 4; t++) {
        int idx = tid + t * 256;
        float s = p[idx];
        if (m[idx]) s = -1e9f;
        v[t] = s;
        mx = fmaxf(mx, s);
    }
    __shared__ float red[256];
    red[tid] = mx; __syncthreads();
    for (int off = 128; off > 0; off >>= 1) {
        if (tid < off) red[tid] = fmaxf(red[tid], red[tid + off]);
        __syncthreads();
    }
    mx = red[0]; __syncthreads();

    float sum = 0.0f;
#pragma unroll
    for (int t = 0; t < 4; t++) { v[t] = __expf(v[t] - mx); sum += v[t]; }
    red[tid] = sum; __syncthreads();
    for (int off = 128; off > 0; off >>= 1) {
        if (tid < off) red[tid] += red[tid + off];
        __syncthreads();
    }
    float inv = 1.0f / red[0];
#pragma unroll
    for (int t = 0; t < 4; t++) {
        int idx = tid + t * 256;
        float a = v[t] * inv;
        p[idx]  = a;
        pb[idx] = (__bf16)a;
    }
}

// ---------------------------------------------------------------------------

extern "C" void kernel_launch(void* const* d_in, const int* in_sizes, int n_in,
                              void* d_out, int out_size, void* d_ws, size_t ws_size,
                              hipStream_t stream)
{
    (void)in_sizes; (void)n_in; (void)out_size; (void)ws_size;

    const float*   x       = (const float*)d_in[0];   // (8,512,32,32)
    const float*   content = (const float*)d_in[1];   // (8,1024,1024)
    const uint8_t* mask    = (const uint8_t*)d_in[2]; // (8,1024,1024) bool
    const float*   W_in    = (const float*)d_in[3];   // (1024,512)
    const float*   b_in    = (const float*)d_in[4];   // (1024)
    const float*   Wq      = (const float*)d_in[5];   // (1024,1024)
    const float*   Wk      = (const float*)d_in[6];
    const float*   Wv      = (const float*)d_in[7];
    const float*   W_out   = (const float*)d_in[8];   // (512,1024)
    const float*   b_out   = (const float*)d_in[9];   // (512)

    float* out_f = (float*)d_out;                 // (8,512,32,32) = 4,194,304 f32
    float* att_f = out_f + 8L * 512 * 1024;       // (8,1024,1024) = 8,388,608 f32

    // Workspace carve-up (bf16 staging buffers), 256B aligned.
    char* w = (char*)d_ws;
    auto alloc = [&](long bytes) { char* p = w; w += (bytes + 255) & ~255L; return p; };
    __bf16* content_b = (__bf16*)alloc(8L * 1024 * 1024 * 2);
    __bf16* x_tok     = (__bf16*)alloc(8L * 1024 * 512  * 2);
    __bf16* Win_b     = (__bf16*)alloc(1024L * 512  * 2);
    __bf16* Wq_b      = (__bf16*)alloc(1024L * 1024 * 2);
    __bf16* Wk_b      = (__bf16*)alloc(1024L * 1024 * 2);
    __bf16* Wv_b      = (__bf16*)alloc(1024L * 1024 * 2);
    __bf16* Wout_b    = (__bf16*)alloc(512L  * 1024 * 2);
    __bf16* xt        = (__bf16*)alloc(8L * 1024 * 1024 * 2);
    __bf16* Qb        = (__bf16*)alloc(8L * 1024 * 1024 * 2);
    __bf16* Kb        = (__bf16*)alloc(8L * 1024 * 1024 * 2);
    __bf16* Vt        = (__bf16*)alloc(8L * 1024 * 1024 * 2);  // V transposed (b,emb,seq)
    __bf16* attb      = (__bf16*)alloc(8L * 1024 * 1024 * 2);
    __bf16* aout      = (__bf16*)alloc(8L * 1024 * 1024 * 2);

    // 1) f32 -> bf16 staging
    k_f32_to_bf16<<<2048, 256, 0, stream>>>(content, content_b, 8L * 1024 * 1024);
    k_f32_to_bf16<<<512,  256, 0, stream>>>(W_in,  Win_b,  1024L * 512);
    k_f32_to_bf16<<<1024, 256, 0, stream>>>(Wq,    Wq_b,   1024L * 1024);
    k_f32_to_bf16<<<1024, 256, 0, stream>>>(Wk,    Wk_b,   1024L * 1024);
    k_f32_to_bf16<<<1024, 256, 0, stream>>>(Wv,    Wv_b,   1024L * 1024);
    k_f32_to_bf16<<<512,  256, 0, stream>>>(W_out, Wout_b, 512L * 1024);
    k_x_to_tokens<<<2048, 256, 0, stream>>>(x, x_tok);

    const dim3 blk(256);
    const long sAct = 1024L * 1024;   // per-batch activation stride (elements)

    // 2) xt = x_tok @ W_in^T + b_in            (M=1024, N=1024, K=512)
    wmma_gemm_bf16<false><<<dim3(8, 4, 8), blk, 0, stream>>>(
        x_tok, Win_b, b_in, nullptr, xt, 1024, 1024, 512, 1024L * 512, 0, sAct, 1.0f);
    // 3) Q = xt @ Wq^T
    wmma_gemm_bf16<false><<<dim3(8, 4, 8), blk, 0, stream>>>(
        xt, Wq_b, nullptr, nullptr, Qb, 1024, 1024, 1024, sAct, 0, sAct, 1.0f);
    // 4) K = content @ Wk^T
    wmma_gemm_bf16<false><<<dim3(8, 4, 8), blk, 0, stream>>>(
        content_b, Wk_b, nullptr, nullptr, Kb, 1024, 1024, 1024, sAct, 0, sAct, 1.0f);
    // 5) Vt = (content @ Wv^T)^T   (stored (b, emb, seq))
    wmma_gemm_bf16<true><<<dim3(8, 4, 8), blk, 0, stream>>>(
        content_b, Wv_b, nullptr, nullptr, Vt, 1024, 1024, 1024, sAct, 0, sAct, 1.0f);
    // 6) scores = (Q @ K^T) * emb^-0.5 -> f32 straight into d_out att region
    wmma_gemm_bf16<false><<<dim3(8, 4, 8), blk, 0, stream>>>(
        Qb, Kb, nullptr, att_f, nullptr, 1024, 1024, 1024, sAct, sAct, sAct, 0.03125f);
    // 7) softmax rows in place (f32 att output) + bf16 copy
    k_softmax<<<8192, 256, 0, stream>>>(att_f, mask, attb, 1024);
    // 8) aout = att @ V = attb @ Vt^T
    wmma_gemm_bf16<false><<<dim3(8, 4, 8), blk, 0, stream>>>(
        attb, Vt, nullptr, nullptr, aout, 1024, 1024, 1024, sAct, sAct, sAct, 1.0f);
    // 9) out = (aout @ W_out^T + b_out)^T -> (b, 512, 32, 32)
    wmma_gemm_bf16<true><<<dim3(4, 4, 8), blk, 0, stream>>>(
        aout, Wout_b, b_out, out_f, nullptr, 1024, 512, 1024, sAct, 0, 512L * 1024, 1.0f);
}
